// BiLSTM_CRF_58729382805598
// MI455X (gfx1250) — matrix operand order repaired and checked by reading, not compiled, not run on hardware
//
#include <hip/hip_runtime.h>
#include <hip/hip_bf16.h>
#include <math.h>

typedef __attribute__((ext_vector_type(16))) __bf16 v16bf;
typedef __attribute__((ext_vector_type(8)))  float  v8f;
typedef __attribute__((ext_vector_type(4)))  unsigned int u32x4;

#define V_  50000
#define T_  20
#define B_  256
#define S_  512
#define E_  256
#define HH_ 128
#define NG_ 512   // 4*HH
#define LSTM_LDS (2 * 64 * HH_ * 2 + NG_ * HH_ * 2)   // 32KB h ping-pong + 128KB Wrec

// ---------------- WMMA helpers (CDNA5 wave32, 16x16x32 bf16, f32 acc) -------

__device__ __forceinline__ v8f wmma_bf16(v16bf a, v16bf b, v8f c) {
  return __builtin_amdgcn_wmma_f32_16x16x32_bf16(false, a, false, b, (short)0, c,
                                                 false, false);
}

// A-fragment (16x32, row-major source, ld in elements).
__device__ __forceinline__ v16bf load_fragA(const __bf16* base, int ld, int lane) {
  int r = lane & 15, h = lane >> 4;
  const __bf16* p0 = base + r * ld + h * 8;
  union { v16bf v; u32x4 q[2]; } f;
  f.q[0] = *(const u32x4*)p0;
  f.q[1] = *(const u32x4*)(p0 + 16);
  return f.v;
}

// B-fragment (32x16) from W^T stored [n][k] row-major (ld in elements).
__device__ __forceinline__ v16bf load_fragB(const __bf16* base, int ld, int lane) {
  int n = lane & 15, h = lane >> 4;
  const __bf16* p0 = base + n * ld + h * 8;
  union { v16bf v; u32x4 q[2]; } f;
  f.q[0] = *(const u32x4*)p0;
  f.q[1] = *(const u32x4*)(p0 + 16);
  return f.v;
}

__device__ __forceinline__ float sigf(float x) {
  return 1.0f / (1.0f + __expf(-x));
}

// ---------------- kernel 0: pack weights ------------------------------------
__global__ void prep_kernel(const float* __restrict__ wih_f, const float* __restrict__ whh_f,
                            const float* __restrict__ bih_f, const float* __restrict__ bhh_f,
                            const float* __restrict__ wih_b, const float* __restrict__ whh_b,
                            const float* __restrict__ bih_b, const float* __restrict__ bhh_b,
                            const float* __restrict__ w_out,
                            __bf16* __restrict__ Wihf, __bf16* __restrict__ Wihb,
                            __bf16* __restrict__ Wrf,  __bf16* __restrict__ Wrb,
                            float* __restrict__ bcf, float* __restrict__ bcb,
                            __bf16* __restrict__ wo) {
  int n = threadIdx.x;                 // 0..511
  for (int k = 0; k < E_; ++k) {
    Wihf[n * E_ + k] = (__bf16)wih_f[n * E_ + k];
    Wihb[n * E_ + k] = (__bf16)wih_b[n * E_ + k];
  }
  for (int k = 0; k < HH_; ++k) {
    Wrf[n * HH_ + k] = (__bf16)whh_f[n * HH_ + k];
    Wrb[n * HH_ + k] = (__bf16)whh_b[n * HH_ + k];
  }
  bcf[n] = bih_f[n] + bhh_f[n];
  bcb[n] = bih_b[n] + bhh_b[n];
  if (n < 32) {
    for (int k = 0; k < E_; ++k)
      wo[n * E_ + k] = (n < T_) ? (__bf16)w_out[n * E_ + k] : (__bf16)0.0f;
  }
}

// ---------------- kernel 1: embedding gather --------------------------------
// x[t][b][e] bf16 (row index = t*256 + b)
__global__ void embed_kernel(const int* __restrict__ token_vec,
                             const float* __restrict__ emb, __bf16* __restrict__ x) {
  const int lane = threadIdx.x & 31;
  const int wave = threadIdx.x >> 5;
  const int row = blockIdx.x * 8 + wave;   // 0..131071
  const int t = row >> 8;
  const int b = row & 255;
  const int tok = token_vec[b * S_ + t];
  const float* e = emb + (size_t)tok * E_ + lane * 8;
  union { __bf16 hv[8]; u32x4 q; } p;
#pragma unroll
  for (int i = 0; i < 8; ++i) p.hv[i] = (__bf16)e[i];
  *(u32x4*)(x + (size_t)row * E_ + lane * 8) = p.q;
}

// ---------------- kernel 2: input projection gx = x_dir @ Wih^T -------------
// Fully parallel WMMA GEMM. Output stored TRANSPOSED as gx[t][n][b] bf16 so the
// LSTM kernel's C-fragment lanes read 8 consecutive b values with one b128 load.
// Backward direction gathers reversed rows per-lane (no xr buffer needed).
__global__ void gx_kernel(const __bf16* __restrict__ x, const int* __restrict__ seq_len,
                          const __bf16* __restrict__ wihf, const __bf16* __restrict__ wihb,
                          __bf16* __restrict__ gxf, __bf16* __restrict__ gxb) {
  const int lane = threadIdx.x & 31;
  const int wave = threadIdx.x >> 5;     // 0..7  -> N base = wave*64
  const int mt = blockIdx.x;             // M tile, 0..8191
  const int dir = blockIdx.y;
  const __bf16* Wih = dir ? wihb : wihf;
  __bf16* gx = dir ? gxb : gxf;
  const int r = lane & 15, h2 = lane >> 4;
  const int m = mt * 16 + r;             // scan-time row: t*256 + b
  const int t = m >> 8;
  const int b = m & 255;
  int ts = t;
  if (dir) { int sl = seq_len[b]; ts = (t < sl) ? (sl - 1 - t) : t; }
  const __bf16* arow = x + ((size_t)ts * B_ + b) * E_;   // per-lane A row
  const int nb = wave * 64;

  v8f acc[4];
#pragma unroll
  for (int c = 0; c < 4; ++c) acc[c] = (v8f){0.f,0.f,0.f,0.f,0.f,0.f,0.f,0.f};

#pragma unroll
  for (int kk = 0; kk < 8; ++kk) {
    union { v16bf v; u32x4 q[2]; } a;
    a.q[0] = *(const u32x4*)(arow + kk * 32 + h2 * 8);
    a.q[1] = *(const u32x4*)(arow + kk * 32 + 16 + h2 * 8);
#pragma unroll
    for (int c = 0; c < 4; ++c) {
      v16bf bm = load_fragB(Wih + (size_t)(nb + c * 16) * E_ + kk * 32, E_, lane);
      acc[c] = wmma_bf16(a.v, bm, acc[c]);
    }
  }
  // transposed store: gx[(t*512 + n)*256 + b]; for fixed lane, v -> consecutive b
  const int tt = mt >> 4;
  const int bb = (mt & 15) * 16 + 8 * h2;
#pragma unroll
  for (int c = 0; c < 4; ++c) {
    int n = nb + c * 16 + r;
    union { __bf16 hv[8]; u32x4 q; } pk;
#pragma unroll
    for (int v = 0; v < 8; ++v) pk.hv[v] = (__bf16)acc[c][v];
    *(u32x4*)(gx + ((size_t)tt * NG_ + n) * B_ + bb) = pk.q;
  }
}

// ---------------- kernel 3: persistent recurrent LSTM -----------------------
// grid = 8 blocks (dir x 4 batch tiles of 64), 512 threads = 16 waves.
// Wave w: row pair rp = w>>3 (M rows rp*32..rp*32+31), j tile jb=(w&7)*16.
// Each wave owns gate tiles {j, j+128, j+256, j+384} for 2 row tiles: i/f/g/o
// element-aligned in C layout -> register-resident cell update.
// Wrec (128KB bf16) + h ping-pong (32KB) live in LDS; only the independent gx
// loads touch global inside the step.
__global__ void lstm_kernel(const __bf16* __restrict__ gxf, const __bf16* __restrict__ gxb,
                            const __bf16* __restrict__ wrf, const __bf16* __restrict__ wrb,
                            const float* __restrict__ bcf, const float* __restrict__ bcb,
                            __bf16* __restrict__ hf, __bf16* __restrict__ hb) {
  extern __shared__ __align__(16) __bf16 sm[];
  __bf16* hsh = sm;                     // 2 * 64 * 128
  __bf16* wsh = sm + 2 * 64 * HH_;      // 512 * 128
  const int dir = blockIdx.x >> 2;
  const int bBase = (blockIdx.x & 3) * 64;
  const __bf16* gx = dir ? gxb : gxf;
  const __bf16* Wr = dir ? wrb : wrf;
  const float* bias = dir ? bcb : bcf;
  __bf16* hout = dir ? hb : hf;

  const int tid = threadIdx.x, lane = tid & 31, wave = tid >> 5;
  const int rp = wave >> 3;             // 0..1
  const int jb = (wave & 7) * 16;       // 0..112
  const int l15 = lane & 15, h2 = lane >> 4;

  {  // stage Wrec into LDS (8192 x b128)
    const u32x4* src = (const u32x4*)Wr;
    u32x4* dst = (u32x4*)wsh;
    for (int i = tid; i < (NG_ * HH_) / 8; i += 512) dst[i] = src[i];
  }
  for (int i = tid; i < 64 * HH_; i += 512) hsh[i] = (__bf16)0.0f;   // h_{-1} = 0
  __syncthreads();

  v8f cst[2];
  cst[0] = (v8f){0.f,0.f,0.f,0.f,0.f,0.f,0.f,0.f};
  cst[1] = cst[0];
  const float bi = bias[0 * HH_ + jb + l15];
  const float bf = bias[1 * HH_ + jb + l15];
  const float bg = bias[2 * HH_ + jb + l15];
  const float bo = bias[3 * HH_ + jb + l15];

  for (int t = 0; t < S_; ++t) {
    const __bf16* hprev = hsh + ((t & 1) ? 64 * HH_ : 0);
    __bf16* hcur        = hsh + ((t & 1) ? 0 : 64 * HH_);

    // gx loads: independent of h, issued first (one b128 per gate per row tile)
    union { __bf16 hv[8]; u32x4 q; } gxv[2][4];
#pragma unroll
    for (int u = 0; u < 2; ++u) {
      const int boff = bBase + rp * 32 + u * 16 + 8 * h2;
#pragma unroll
      for (int g = 0; g < 4; ++g)
        gxv[u][g].q = *(const u32x4*)(gx + ((size_t)t * NG_ + g * HH_ + jb + l15) * B_ + boff);
    }
    if (t + 1 < S_) {    // prefetch next step's gx (-> global_prefetch_b8)
#pragma unroll
      for (int g = 0; g < 4; ++g)
        __builtin_prefetch(gx + ((size_t)(t + 1) * NG_ + g * HH_ + jb + l15) * B_ +
                               bBase + rp * 32 + 8 * h2, 0, 1);
    }

    v8f acc[2][4];
#pragma unroll
    for (int u = 0; u < 2; ++u)
#pragma unroll
      for (int g = 0; g < 4; ++g) acc[u][g] = (v8f){0.f,0.f,0.f,0.f,0.f,0.f,0.f,0.f};

    // recurrent GEMM: K = 128, both operands from LDS
#pragma unroll
    for (int kk = 0; kk < 4; ++kk) {
      v16bf a0 = load_fragA(hprev + (rp * 32) * HH_ + kk * 32, HH_, lane);
      v16bf a1 = load_fragA(hprev + (rp * 32 + 16) * HH_ + kk * 32, HH_, lane);
#pragma unroll
      for (int g = 0; g < 4; ++g) {
        v16bf bm = load_fragB(wsh + (size_t)(g * HH_ + jb) * HH_ + kk * 32, HH_, lane);
        acc[0][g] = wmma_bf16(a0, bm, acc[0][g]);
        acc[1][g] = wmma_bf16(a1, bm, acc[1][g]);
      }
    }

    // cell update (register-resident)
#pragma unroll
    for (int u = 0; u < 2; ++u) {
#pragma unroll
      for (int v = 0; v < 8; ++v) {
        float iv = sigf(acc[u][0][v] + (float)gxv[u][0].hv[v] + bi);
        float fv = sigf(acc[u][1][v] + (float)gxv[u][1].hv[v] + bf);
        float gv = tanhf(acc[u][2][v] + (float)gxv[u][2].hv[v] + bg);
        float ov = sigf(acc[u][3][v] + (float)gxv[u][3].hv[v] + bo);
        float cv = fv * cst[u][v] + iv * gv;
        cst[u][v] = cv;
        float hv = ov * tanhf(cv);
        __bf16 hw = (__bf16)hv;
        int brow = rp * 32 + u * 16 + v + h2 * 8;   // C layout: M = v + 8*(lane>>4)
        int col  = jb + l15;
        hcur[brow * HH_ + col] = hw;
        hout[((size_t)t * B_ + bBase + brow) * HH_ + col] = hw;
      }
    }
    __syncthreads();
  }
}

// ---------------- kernel 4: feats = [h_f | h_b[rev]] @ w_out^T + b_out ------
__global__ void feats_kernel(const __bf16* __restrict__ hf, const __bf16* __restrict__ hb,
                             const __bf16* __restrict__ wo, const float* __restrict__ b_out,
                             const int* __restrict__ seq_len, float* __restrict__ feats) {
  const int lane = threadIdx.x & 31;
  const int wave = threadIdx.x >> 5;
  const int mt = blockIdx.x * 8 + wave;      // M tile, 0..8191 (m = b*512 + t)
  const int r = lane & 15, h2 = lane >> 4;
  const int m = mt * 16 + r;
  const int b = m >> 9;
  const int t = m & 511;
  const int sl = seq_len[b];
  const int rv = (t < sl) ? (sl - 1 - t) : t;
  const __bf16* rowf = hf + ((size_t)t  * B_ + b) * HH_;
  const __bf16* rowb = hb + ((size_t)rv * B_ + b) * HH_;

  v8f acc0 = {0.f,0.f,0.f,0.f,0.f,0.f,0.f,0.f};
  v8f acc1 = {0.f,0.f,0.f,0.f,0.f,0.f,0.f,0.f};
#pragma unroll
  for (int kk = 0; kk < 8; ++kk) {
    int k0 = kk * 32 + h2 * 8;
    int k1 = k0 + 16;
    union { v16bf v; u32x4 q[2]; } a;
    a.q[0] = *(const u32x4*)((k0 < HH_) ? (rowf + k0) : (rowb + (k0 - HH_)));
    a.q[1] = *(const u32x4*)((k1 < HH_) ? (rowf + k1) : (rowb + (k1 - HH_)));
    v16bf b0 = load_fragB(wo + 0       + kk * 32, E_, lane);
    v16bf b1 = load_fragB(wo + 16 * E_ + kk * 32, E_, lane);
    acc0 = wmma_bf16(a.v, b0, acc0);
    acc1 = wmma_bf16(a.v, b1, acc1);
  }
#pragma unroll
  for (int v = 0; v < 8; ++v) {
    int mm = mt * 16 + v + h2 * 8;
    int n0 = r;
    int n1 = 16 + r;
    feats[(size_t)mm * T_ + n0] = acc0[v] + b_out[n0];
    if (n1 < T_) feats[(size_t)mm * T_ + n1] = acc1[v] + b_out[n1];
  }
}

// ---------------- kernel 5: Viterbi forward (wave per batch row) ------------
__global__ void viterbi_kernel(const float* __restrict__ feats, const int* __restrict__ token_vec,
                               const float* __restrict__ trans,
                               float* __restrict__ dfin, int* __restrict__ bps) {
  __shared__ float tr[T_ * T_];
  const int tid = threadIdx.x;
  if (tid < T_ * T_) tr[tid] = trans[tid];
  __syncthreads();
  const int lane = tid & 31;
  const int wave = tid >> 5;
  const int b = blockIdx.x * 8 + wave;
  const int j = lane;
  const int jc = (j < T_) ? j : 0;

  float d = (j < T_) ? feats[((size_t)b * S_) * T_ + j] : -3.0e38f;
  for (int t = 1; t < S_; ++t) {
    const bool m = token_vec[b * S_ + t] != 0;
    const float ft = (j < T_) ? feats[((size_t)b * S_ + t) * T_ + j] : 0.0f;
    float best = -3.0e38f;
    int arg = 0;
#pragma unroll
    for (int i = 0; i < T_; ++i) {
      float di = __shfl(d, i, 32);
      float v = di + tr[i * T_ + jc] + ft;
      if (v > best) { best = v; arg = i; }
    }
    int bp = m ? arg : j;
    if (m) d = best;
    if (j < T_) bps[(size_t)((t - 1) * B_ + b) * T_ + j] = bp;
  }
  if (j < T_) dfin[b * T_ + j] = d;
}

// ---------------- kernel 6: score + backtrack -------------------------------
__global__ void backtrack_kernel(const float* __restrict__ dfin, const int* __restrict__ bps,
                                 float* __restrict__ out) {
  const int b = threadIdx.x;
  float best = -3.0e38f;
  int last = 0;
  for (int j = 0; j < T_; ++j) {
    float v = dfin[b * T_ + j];
    if (v > best) { best = v; last = j; }
  }
  out[b] = best;
  float* tout = out + B_;
  int tag = last;
  tout[(size_t)b * S_ + (S_ - 1)] = (float)tag;
  for (int k = S_ - 2; k >= 0; --k) {
    tag = bps[(size_t)(k * B_ + b) * T_ + tag];
    tout[(size_t)b * S_ + k] = (float)tag;
  }
}

// ---------------- launch -----------------------------------------------------
extern "C" void kernel_launch(void* const* d_in, const int* in_sizes, int n_in,
                              void* d_out, int out_size, void* d_ws, size_t ws_size,
                              hipStream_t stream) {
  const int*   token_vec = (const int*)d_in[0];
  const int*   seq_len   = (const int*)d_in[2];
  const float* embedding = (const float*)d_in[3];
  const float* wih_f = (const float*)d_in[4];
  const float* whh_f = (const float*)d_in[5];
  const float* bih_f = (const float*)d_in[6];
  const float* bhh_f = (const float*)d_in[7];
  const float* wih_b = (const float*)d_in[8];
  const float* whh_b = (const float*)d_in[9];
  const float* bih_b = (const float*)d_in[10];
  const float* bhh_b = (const float*)d_in[11];
  const float* w_out = (const float*)d_in[12];
  const float* b_out = (const float*)d_in[13];
  const float* trans = (const float*)d_in[14];

  char* w = (char*)d_ws;
  size_t o = 0;
  auto alloc = [&](size_t bytes) {
    void* p = (void*)(w + o);
    o = (o + bytes + 255) & ~(size_t)255;
    return p;
  };
  __bf16* x    = (__bf16*)alloc((size_t)S_ * B_ * E_ * 2);    // 64 MB
  __bf16* gxf  = (__bf16*)alloc((size_t)S_ * NG_ * B_ * 2);   // 128 MB  [t][n][b]
  __bf16* gxb  = (__bf16*)alloc((size_t)S_ * NG_ * B_ * 2);   // 128 MB
  __bf16* hfst = (__bf16*)alloc((size_t)S_ * B_ * HH_ * 2);   // 32 MB   [t][b][j]
  __bf16* hbst = (__bf16*)alloc((size_t)S_ * B_ * HH_ * 2);   // 32 MB
  __bf16* Wihf = (__bf16*)alloc((size_t)NG_ * E_ * 2);
  __bf16* Wihb = (__bf16*)alloc((size_t)NG_ * E_ * 2);
  __bf16* Wrf  = (__bf16*)alloc((size_t)NG_ * HH_ * 2);
  __bf16* Wrb  = (__bf16*)alloc((size_t)NG_ * HH_ * 2);
  float*  bcf  = (float*)alloc((size_t)NG_ * 4);
  float*  bcb  = (float*)alloc((size_t)NG_ * 4);
  __bf16* wo   = (__bf16*)alloc((size_t)32 * E_ * 2);
  float*  feats = (float*)alloc((size_t)B_ * S_ * T_ * 4);    // 10 MB
  float*  dfin  = (float*)alloc((size_t)B_ * T_ * 4);
  int*    bps   = (int*)alloc((size_t)(S_ - 1) * B_ * T_ * 4);

  hipFuncSetAttribute((const void*)lstm_kernel,
                      hipFuncAttributeMaxDynamicSharedMemorySize, LSTM_LDS);

  prep_kernel<<<1, 512, 0, stream>>>(wih_f, whh_f, bih_f, bhh_f,
                                     wih_b, whh_b, bih_b, bhh_b, w_out,
                                     Wihf, Wihb, Wrf, Wrb, bcf, bcb, wo);
  embed_kernel<<<(S_ * B_) / 8, 256, 0, stream>>>(token_vec, embedding, x);
  gx_kernel<<<dim3(S_ * B_ / 16, 2), 256, 0, stream>>>(x, seq_len, Wihf, Wihb, gxf, gxb);
  lstm_kernel<<<8, 512, LSTM_LDS, stream>>>(gxf, gxb, Wrf, Wrb, bcf, bcb, hfst, hbst);
  feats_kernel<<<(S_ * B_ / 16) / 8, 256, 0, stream>>>(hfst, hbst, wo, b_out, seq_len, feats);
  viterbi_kernel<<<B_ / 8, 256, 0, stream>>>(feats, token_vec, trans, dfin, bps);
  backtrack_kernel<<<1, B_, 0, stream>>>(dfin, bps, (float*)d_out);
}
